// NetTGCNBasic_7164005450119
// MI455X (gfx1250) — matrix operand order repaired and checked by reading, not compiled, not run on hardware
//
#include <hip/hip_runtime.h>

// Problem dims (fixed by reference)
#define Bb 32
#define Nn 16384
#define Hh 15
#define Kk 25
#define Gg 64
#define Cc 6
#define BH (Bb * Hh)          // 480
#define NG (Nn * Gg)          // 1048576
#define CHUNK 1024
#define NCHUNK (NG / CHUNK)   // 1024
#define NT (Nn / 16)          // 1024 node tiles
#define CONTRACT_BLOCKS ((NT * Bb) / 8)   // 8 waves per 256-thread block

typedef __attribute__((ext_vector_type(16))) __bf16 v16bf;
typedef __attribute__((ext_vector_type(8)))  float  v8f;

static __device__ __forceinline__ unsigned short f2bf_bits(float f) {
  unsigned u = __builtin_bit_cast(unsigned, f);
  unsigned r = (u + 0x7FFFu + ((u >> 16) & 1u)) >> 16;   // RNE
  return (unsigned short)r;
}
static __device__ __forceinline__ __bf16 f2bf(float f) {
  unsigned short h = f2bf_bits(f);
  return __builtin_bit_cast(__bf16, h);
}

// ---------------------------------------------------------------------------
// X0[n, b*H + h] = x[b, n, 0, h]
__global__ void build_x0_kernel(const float* __restrict__ x, float* __restrict__ X0) {
  int idx = blockIdx.x * blockDim.x + threadIdx.x;
  if (idx >= Nn * BH) return;
  int n = idx / BH;
  int r = idx - n * BH;
  int b = r / Hh;
  int h = r - b * Hh;
  X0[idx] = x[((size_t)b * Nn + n) * Hh + h];
}

// ---------------------------------------------------------------------------
// Xnext = (mode==1) ? -Xkm2 : 0      (pre-bias for Xk = 2*L@Xkm1 - Xkm2)
__global__ void spmm_init_kernel(float* __restrict__ Xnext,
                                 const float* __restrict__ Xkm2, int mode) {
  int idx = blockIdx.x * blockDim.x + threadIdx.x;
  if (idx >= Nn * BH) return;
  Xnext[idx] = mode ? -Xkm2[idx] : 0.0f;
}

// ---------------------------------------------------------------------------
// One wave per edge: Xout[src, :] += alpha * w_e * Xin[dst, :]   (480 columns)
// X fits in L2 (31.5 MB << 192 MB): gathers + f32 atomics stay on-chip.
__global__ void spmm_scatter_kernel(const int* __restrict__ esrc,
                                    const int* __restrict__ edst,
                                    const float* __restrict__ ew,
                                    const float* __restrict__ Xin,
                                    float* __restrict__ Xout,
                                    float alpha, int nedges) {
  int gtid = blockIdx.x * blockDim.x + threadIdx.x;
  int e    = gtid >> 5;
  int lane = gtid & 31;
  if (e >= nedges) return;
  int s = esrc[e];
  int d = edst[e];
  float a = alpha * ew[e];
  const float* __restrict__ xin  = Xin  + (size_t)d * BH;
  float* __restrict__       xout = Xout + (size_t)s * BH;
#pragma unroll
  for (int c = 0; c < BH / 32; ++c) {   // 15 coalesced column groups
    int col = lane + 32 * c;
    unsafeAtomicAdd(&xout[col], a * xin[col]);
  }
}

// ---------------------------------------------------------------------------
// out_acc[b, n, g] (+)= sum_{k<nk} sum_h Xk[n, b*15+h] * W[k0+k, h, g]
// One wave owns a (16-node tile, b); keeps FOUR 16x16 f32 C tiles in registers
// across all nk slices (4 WMMAs per slice), writes out_acc exactly once.
// W slices staged in LDS as bf16, layout [k][g][16 rows] -> one contiguous
// 32-B read per lane per B tile; a shared zeroed block serves the K=16..31
// padding lanes (address select, no exec divergence anywhere).
__global__ void contract_wmma_kernel(const float* __restrict__ Xbase,
                                     size_t xstride,                  // elems between k-slices
                                     const float* __restrict__ Wbase, // W + k0*H*G
                                     float* __restrict__ out_acc,     // [B, N, G]
                                     int nk, int first) {
  extern __shared__ __align__(64) unsigned short lds_w[];  // (nk+1)*16*64 bf16

  // ---- cooperative LDS fill: slice k row-major [g][r], rows >= H zeroed,
  // ---- plus one fully-zero block at offset nk*1024 for the padding lanes.
  int t = threadIdx.x;
  int total = nk * 1024;
  for (int i = t; i < total + 1024; i += 256) {
    unsigned short v = 0;
    if (i < total) {
      int k = i >> 10;
      int g = (i >> 4) & 63;
      int r = i & 15;
      if (r < Hh) v = f2bf_bits(Wbase[(size_t)k * (Hh * Gg) + r * Gg + g]);
    }
    lds_w[i] = v;
  }
  __syncthreads();

  int wave = blockIdx.x * 8 + (t >> 5);     // grid sized exactly: no guard needed
  int lane = t & 31;
  int nt = wave % NT;
  int b  = wave / NT;
  int mr   = lane & 15;
  int half = lane >> 4;

  // B-tile LDS base: half=0 lanes read slice k, half=1 lanes read zero block
  const unsigned short* __restrict__ zblock = lds_w + (size_t)nk * 1024;

  // C accumulators: rows m = half*8 + r, col = gt*16 + mr
  v8f c[4];
  float* __restrict__ obase =
      out_acc + (size_t)b * NG + (size_t)(nt * 16 + half * 8) * Gg + mr;
#pragma unroll
  for (int gt = 0; gt < 4; ++gt) {
#pragma unroll
    for (int r = 0; r < 8; ++r)
      c[gt][r] = first ? 0.0f : obase[(size_t)r * Gg + gt * 16];
  }

  // A row pointer: row n = nt*16+mr, cols b*15 + e
  const float* __restrict__ xrow =
      Xbase + (size_t)(nt * 16 + mr) * BH + b * Hh;

  for (int k = 0; k < nk; ++k, xrow += xstride) {
    // ---- A tile: 16x32 bf16 (lanes0-15: K=0..15 ; lanes16-31: K=16..31 zero padding)
    v16bf a;
#pragma unroll
    for (int e = 0; e < 16; ++e) {
      bool valid = (half == 0) && (e < Hh);
      float v = xrow[valid ? e : 0];             // clamped in-range unconditional load
      a[e] = f2bf(valid ? v : 0.0f);
    }

    const unsigned short* __restrict__ lbase =
        half ? zblock : (lds_w + (size_t)k * 1024);
#pragma unroll
    for (int gt = 0; gt < 4; ++gt) {
      // ---- B tile: 32x16 bf16; lane reads 16 contiguous bf16 (32 B) from LDS
      v16bf bm = *(const v16bf*)(lbase + (size_t)(gt * 16 + mr) * 16);
      c[gt] = __builtin_amdgcn_wmma_f32_16x16x32_bf16(
          false, a, false, bm, (short)0, c[gt], false, false);
    }
  }

#pragma unroll
  for (int gt = 0; gt < 4; ++gt) {
#pragma unroll
    for (int r = 0; r < 8; ++r)
      obase[(size_t)r * Gg + gt * 16] = c[gt][r];
  }
}

// ---------------------------------------------------------------------------
// FC pass 1: per (b, chunk) block, partial[c] = sum_j relu(out+b_conv)*W_fc[c,j]
__global__ void fc_partial_kernel(const float* __restrict__ out_acc,
                                  const float* __restrict__ b_conv,
                                  const float* __restrict__ W_fc,
                                  float* __restrict__ partial) {
  int blk   = blockIdx.x;
  int b     = blk / NCHUNK;
  int chunk = blk - b * NCHUNK;
  int t = threadIdx.x;

  float acc[Cc];
#pragma unroll
  for (int c = 0; c < Cc; ++c) acc[c] = 0.0f;

  for (int i = t; i < CHUNK; i += 256) {
    int j = chunk * CHUNK + i;                 // inner index into N*G
    float v = out_acc[(size_t)b * NG + j] + b_conv[j & (Gg - 1)];
    v = fmaxf(v, 0.0f);
#pragma unroll
    for (int c = 0; c < Cc; ++c) acc[c] += v * W_fc[(size_t)c * NG + j];
  }
#pragma unroll
  for (int off = 16; off > 0; off >>= 1) {
#pragma unroll
    for (int c = 0; c < Cc; ++c) acc[c] += __shfl_down(acc[c], off, 32);
  }
  __shared__ float red[8][Cc];
  int wid = t >> 5, lane = t & 31;
  if (lane == 0) {
#pragma unroll
    for (int c = 0; c < Cc; ++c) red[wid][c] = acc[c];
  }
  __syncthreads();
  if (t < Cc) {
    float s = 0.0f;
#pragma unroll
    for (int w = 0; w < 8; ++w) s += red[w][t];
    partial[((size_t)b * NCHUNK + chunk) * 8 + t] = s;
  }
}

// ---------------------------------------------------------------------------
// FC pass 2: reduce chunks, add bias, log_softmax over C=6, write [B,C]
__global__ void fc_finish_kernel(const float* __restrict__ partial,
                                 const float* __restrict__ b_fc,
                                 float* __restrict__ out) {
  __shared__ float logits[Bb][Cc];
  int t = threadIdx.x;
  if (t < Bb * Cc) {
    int b = t / Cc, c = t - b * Cc;
    float s = b_fc[c];
    for (int ch = 0; ch < NCHUNK; ++ch)
      s += partial[((size_t)b * NCHUNK + ch) * 8 + c];
    logits[b][c] = s;
  }
  __syncthreads();
  if (t < Bb * Cc) {
    int b = t / Cc, c = t - b * Cc;
    float m = logits[b][0];
#pragma unroll
    for (int j = 1; j < Cc; ++j) m = fmaxf(m, logits[b][j]);
    float se = 0.0f;
#pragma unroll
    for (int j = 0; j < Cc; ++j) se += __expf(logits[b][j] - m);
    out[b * Cc + c] = (logits[b][c] - m) - __logf(se);
  }
}

// ---------------------------------------------------------------------------
extern "C" void kernel_launch(void* const* d_in, const int* in_sizes, int n_in,
                              void* d_out, int out_size, void* d_ws, size_t ws_size,
                              hipStream_t stream) {
  const float* x      = (const float*)d_in[0];
  const float* ew     = (const float*)d_in[1];
  const float* W      = (const float*)d_in[2];   // [K, H, 1, G]
  const float* b_conv = (const float*)d_in[3];
  const float* W_fc   = (const float*)d_in[4];   // [C, N*G]
  const float* b_fc   = (const float*)d_in[5];
  const int*   esrc   = (const int*)d_in[6];
  const int*   edst   = (const int*)d_in[7];
  float*       out    = (float*)d_out;
  const int    E      = in_sizes[1];

  const size_t xsz_e  = (size_t)Nn * BH;            // 7.86M elems, 31.5 MB
  const size_t acc_e  = (size_t)Bb * NG;            // 32M elems, 128 MB
  const size_t part_e = (size_t)Bb * NCHUNK * 8;

  const int nElem   = Nn * BH;
  const int eBlocks = (nElem + 255) / 256;
  const int sBlocks = (E * 32 + 255) / 256;         // one wave per edge

  const size_t need_fused = (Kk * xsz_e + acc_e + part_e) * sizeof(float);

  if (ws_size >= need_fused) {
    // ---------------- fused path: store all K slices, contract once ----------
    float* Xall    = (float*)d_ws;
    float* out_acc = Xall + Kk * xsz_e;
    float* partial = out_acc + acc_e;

    build_x0_kernel<<<eBlocks, 256, 0, stream>>>(x, Xall);
    // X1 = L @ X0
    spmm_init_kernel<<<eBlocks, 256, 0, stream>>>(Xall + xsz_e, Xall, 0);
    spmm_scatter_kernel<<<sBlocks, 256, 0, stream>>>(esrc, edst, ew,
        Xall, Xall + xsz_e, 1.0f, E);
    // Xk = 2*L@Xkm1 - Xkm2
    for (int k = 2; k < Kk; ++k) {
      float* Xk   = Xall + (size_t)k * xsz_e;
      float* Xkm1 = Xall + (size_t)(k - 1) * xsz_e;
      float* Xkm2 = Xall + (size_t)(k - 2) * xsz_e;
      spmm_init_kernel<<<eBlocks, 256, 0, stream>>>(Xk, Xkm2, 1);
      spmm_scatter_kernel<<<sBlocks, 256, 0, stream>>>(esrc, edst, ew, Xkm1, Xk, 2.0f, E);
    }
    // one pass: 100 WMMAs per wave, out_acc written once (no RMW)
    size_t lds_bytes = (size_t)(Kk + 1) * 1024 * sizeof(unsigned short);  // 53 KB
    contract_wmma_kernel<<<CONTRACT_BLOCKS, 256, lds_bytes, stream>>>(
        Xall, xsz_e, W, out_acc, Kk, 1);

    fc_partial_kernel<<<Bb * NCHUNK, 256, 0, stream>>>(out_acc, b_conv, W_fc, partial);
    fc_finish_kernel<<<1, 256, 0, stream>>>(partial, b_fc, out);
  } else {
    // ---------------- fallback: ping-pong slices, contract per k -------------
    char* ws = (char*)d_ws;
    float* Xa      = (float*)ws;  ws += xsz_e * sizeof(float);
    float* Xb      = (float*)ws;  ws += xsz_e * sizeof(float);
    float* Xc      = (float*)ws;  ws += xsz_e * sizeof(float);
    float* out_acc = (float*)ws;  ws += acc_e * sizeof(float);
    float* partial = (float*)ws;

    size_t lds1 = 2 * 1024 * sizeof(unsigned short);

    build_x0_kernel<<<eBlocks, 256, 0, stream>>>(x, Xa);
    contract_wmma_kernel<<<CONTRACT_BLOCKS, 256, lds1, stream>>>(
        Xa, 0, W, out_acc, 1, 1);

    spmm_init_kernel<<<eBlocks, 256, 0, stream>>>(Xb, Xa, 0);
    spmm_scatter_kernel<<<sBlocks, 256, 0, stream>>>(esrc, edst, ew, Xa, Xb, 1.0f, E);
    contract_wmma_kernel<<<CONTRACT_BLOCKS, 256, lds1, stream>>>(
        Xb, 0, W + Hh * Gg, out_acc, 1, 0);

    float *Xkm2 = Xa, *Xkm1 = Xb, *Xk = Xc;
    for (int k = 2; k < Kk; ++k) {
      spmm_init_kernel<<<eBlocks, 256, 0, stream>>>(Xk, Xkm2, 1);
      spmm_scatter_kernel<<<sBlocks, 256, 0, stream>>>(esrc, edst, ew, Xkm1, Xk, 2.0f, E);
      contract_wmma_kernel<<<CONTRACT_BLOCKS, 256, lds1, stream>>>(
          Xk, 0, W + (size_t)k * Hh * Gg, out_acc, 1, 0);
      float* tmp = Xkm2; Xkm2 = Xkm1; Xkm1 = Xk; Xk = tmp;
    }

    fc_partial_kernel<<<Bb * NCHUNK, 256, 0, stream>>>(out_acc, b_conv, W_fc, partial);
    fc_finish_kernel<<<1, 256, 0, stream>>>(partial, b_fc, out);
  }
}
